// MyModel_61933428411933
// MI455X (gfx1250) — compile-verified, matrix operand order checked by reading, not run on hardware
//
#include <hip/hip_runtime.h>

// ---------------------------------------------------------------------------
// LSTM forward, persistent-kernel, CDNA5 (gfx1250) WMMA f16 -> f32.
//
//   gates[t] = [x_t | h_{t-1}] @ [W_ih | W_hh]^T + (b_ih + b_hh)
//   i,f,o = sigmoid;  g = tanh;  c = f*c + i*g;  h = o*tanh(c)
//
// B=32 rows (2 M-tiles), 4H=4096 gate cols (256 N-tiles), K=E+H=2048.
// 64 persistent blocks; block owns 16 h-columns (=> 4 gate N-tiles).
// 8 waves/block = (2 M-tiles) x (4 K-quarters); each wave keeps 4 independent
// WMMA accumulator chains (one per gate) -> good ILP on the serial path.
// One agent-scope atomic grid barrier per timestep; h double-buffered in f16.
// ---------------------------------------------------------------------------

typedef __attribute__((ext_vector_type(16))) _Float16 v16h;
typedef __attribute__((ext_vector_type(8)))  _Float16 v8h;
typedef __attribute__((ext_vector_type(8)))  float    v8f;
typedef __attribute__((ext_vector_type(4)))  float    v4f;

#define B_    32
#define T_    2048
#define E_    1024
#define H_    1024
#define K2_   2048          // E + H
#define NWG   64            // persistent workgroups (must all be co-resident)

// Workspace layout (bytes). Total ~17.1 MB.
#define OFF_W     ((size_t)0)                       // 4096*2048 f16 = 16 MB
#define OFF_BIAS  ((size_t)16777216)                // 4096 f32
#define OFF_H0    ((size_t)16793600)                // 32*1024 f16
#define OFF_H1    ((size_t)16924672)                // 32*1024 f16
#define OFF_BAR   ((size_t)17055744)                // barrier counter

// ---------------------------------------------------------------------------
// Pack [W_ih | W_hh] (fp32, row-major [4096][1024] each) into the per-lane
// WMMA B-fragment layout for v_wmma_f32_16x16x32_f16, converted to f16.
// Assumed B layout mirrors the documented 16-bit A layout:
//   lane<16 : n = lane,    K chunks {kt*32+0..7,  kt*32+16..23}
//   lane>=16: n = lane-16, K chunks {kt*32+8..15, kt*32+24..31}
// v16h element j -> K = kbase + j + (j>=8 ? 8 : 0).
// Storage: Wpk[((ntile*64 + kt)*32 + lane)*16 + j], ntile = g_col_tile (256).
// ---------------------------------------------------------------------------
__global__ void pack_weights_kernel(const float* __restrict__ Wih,
                                    const float* __restrict__ Whh,
                                    _Float16* __restrict__ Wpk) {
    int tidg = blockIdx.x * 256 + threadIdx.x;   // 256 ntiles * 64 kt * 32 lanes
    int lane = tidg & 31;
    int kt   = (tidg >> 5) & 63;
    int nt   = tidg >> 11;                       // 0..255
    int n     = lane & 15;
    int g     = nt * 16 + n;                     // gate row 0..4095
    int kbase = kt * 32 + ((lane < 16) ? 0 : 8);
    v16h v;
#pragma unroll
    for (int j = 0; j < 16; ++j) {
        int k = kbase + j + ((j >= 8) ? 8 : 0);
        float w = (k < E_) ? Wih[(size_t)g * E_ + k]
                           : Whh[(size_t)g * H_ + (k - E_)];
        v[j] = (_Float16)w;
    }
    *(v16h*)(Wpk + ((size_t)(nt * 64 + kt) * 32 + lane) * 16) = v;
}

__global__ void sum_bias_kernel(const float* __restrict__ bih,
                                const float* __restrict__ bhh,
                                float* __restrict__ bias) {
    int i = blockIdx.x * 256 + threadIdx.x;      // 4096 threads
    bias[i] = bih[i] + bhh[i];
}

// ---------------------------------------------------------------------------
// Persistent LSTM kernel.
// ---------------------------------------------------------------------------
__launch_bounds__(256, 1)
__global__ void lstm_persistent_kernel(const float*    __restrict__ x,     // [B][T][E]
                                       const _Float16* __restrict__ Wpk,
                                       const float*    __restrict__ bias,  // [4096]
                                       _Float16* __restrict__ hb0,
                                       _Float16* __restrict__ hb1,
                                       unsigned* __restrict__ bar,
                                       float*    __restrict__ out) {
    __shared__ float lds_g[4][B_ * 16];   // per-gate pre-activations (bias-seeded)
    __shared__ float lds_c[B_ * 16];      // persistent cell state for this block

    const int tid  = threadIdx.x;
    const int lane = tid & 31;
    const int wave = tid >> 5;
    const int mt   = wave & 1;            // M tile: rows [mt*16, mt*16+16)
    const int ks   = wave >> 1;           // K quarter: [ks*512, ks*512+512)
    const int blk  = blockIdx.x;          // owns h columns [blk*16, blk*16+16)

    for (int e = tid; e < B_ * 16; e += 256) lds_c[e] = 0.0f;

    // A-fragment geometry (16-bit A layout, 16x32 tile)
    const int arow = mt * 16 + (lane & 15);
    const int koff = (lane < 16) ? 0 : 8;

    // B-fragment base pointers, one per gate, for this block's column tile.
    const _Float16* wp[4];
#pragma unroll
    for (int g = 0; g < 4; ++g) {
        size_t ntile = (size_t)(g * 64 + blk);
        wp[g] = Wpk + (ntile * 64 + (size_t)ks * 16) * (32 * 16) + (size_t)lane * 16;
    }

    const int mbase = mt * 16 + ((lane >> 4) << 3);   // C-layout row base
    const int ncol  = lane & 15;

#pragma unroll 1
    for (int t = 0; t < T_; ++t) {
        // Seed gate accumulators in LDS with the combined bias.
        for (int e = tid; e < 4 * B_ * 16; e += 256) {
            ((float*)lds_g)[e] = bias[(e >> 9) * H_ + blk * 16 + (e & 15)];
        }
        __syncthreads();

        v8f acc[4] = {};   // 4 independent accumulation chains (one per gate)

        if (ks < 2) {
            // ---- K in x-region: load fp32, convert to f16 fragments --------
            const float* xr = x + ((size_t)arow * T_ + t) * E_ + ks * 512;
#pragma unroll
            for (int i = 0; i < 16; ++i) {
                int k0 = i * 32 + koff;
                v4f f0 = *(const v4f*)(xr + k0);
                v4f f1 = *(const v4f*)(xr + k0 + 4);
                v4f f2 = *(const v4f*)(xr + k0 + 16);
                v4f f3 = *(const v4f*)(xr + k0 + 20);
                v16h a;
#pragma unroll
                for (int j = 0; j < 4; ++j) {
                    a[j]      = (_Float16)f0[j];
                    a[4 + j]  = (_Float16)f1[j];
                    a[8 + j]  = (_Float16)f2[j];
                    a[12 + j] = (_Float16)f3[j];
                }
#pragma unroll
                for (int g = 0; g < 4; ++g) {
                    v16h b = *(const v16h*)(wp[g] + (size_t)i * (32 * 16));
                    acc[g] = __builtin_amdgcn_wmma_f32_16x16x32_f16(
                        false, a, false, b, (short)0, acc[g], false, false);
                }
            }
        } else {
            // ---- K in h-region: h already stored as f16 --------------------
            const _Float16* hr = ((t & 1) ? hb1 : hb0)
                               + (size_t)arow * H_ + (size_t)(ks - 2) * 512;
#pragma unroll
            for (int i = 0; i < 16; ++i) {
                int k0 = i * 32 + koff;
                v8h c0 = *(const v8h*)(hr + k0);
                v8h c1 = *(const v8h*)(hr + k0 + 16);
                v16h a;
#pragma unroll
                for (int j = 0; j < 8; ++j) { a[j] = c0[j]; a[8 + j] = c1[j]; }
#pragma unroll
                for (int g = 0; g < 4; ++g) {
                    v16h b = *(const v16h*)(wp[g] + (size_t)i * (32 * 16));
                    acc[g] = __builtin_amdgcn_wmma_f32_16x16x32_f16(
                        false, a, false, b, (short)0, acc[g], false, false);
                }
            }
        }

        // Reduce K-split partials into LDS (ds_add_f32 atomics).
#pragma unroll
        for (int g = 0; g < 4; ++g) {
            float* dstg = &lds_g[g][0];
#pragma unroll
            for (int r = 0; r < 8; ++r)
                atomicAdd(&dstg[(mbase + r) * 16 + ncol], acc[g][r]);
        }
        __syncthreads();

        // Elementwise LSTM cell update for this block's 32x16 slice.
        _Float16* hn = ((t + 1) & 1) ? hb1 : hb0;
        const float* lg = (const float*)lds_g;
        for (int e = tid; e < B_ * 16; e += 256) {
            int brow = e >> 4;
            int n    = e & 15;
            float gi = lg[e];
            float gf = lg[512 + e];
            float gg = lg[1024 + e];
            float go = lg[1536 + e];
            float iv = 1.0f / (1.0f + __expf(-gi));
            float fv = 1.0f / (1.0f + __expf(-gf));
            float gv = tanhf(gg);
            float ov = 1.0f / (1.0f + __expf(-go));
            float cv = fv * lds_c[e] + iv * gv;
            float hv = ov * tanhf(cv);
            lds_c[e] = cv;
            int col = blk * 16 + n;
            hn[(size_t)brow * H_ + col] = (_Float16)hv;
            if (t == T_ - 1) {
                out[(size_t)brow * H_ + col] = hv;                       // h_t
                out[(size_t)B_ * H_ + (size_t)brow * H_ + col] = cv;     // c_t
            }
        }

        // Prefetch next timestep's x slice (disjoint 1/NWG share per block).
        if (t + 1 < T_) {
            const float* pf = x + ((size_t)(tid >> 3) * T_ + (t + 1)) * E_
                            + blk * 16 + (tid & 7) * 2;
            __builtin_prefetch(pf, 0, 3);

            // ---- device-wide barrier (monotonic counter, agent scope) -----
            __threadfence();          // release this wave's h stores to device
            __syncthreads();
            if (tid == 0) {
                __hip_atomic_fetch_add(bar, 1u, __ATOMIC_RELEASE,
                                       __HIP_MEMORY_SCOPE_AGENT);
                unsigned target = (unsigned)NWG * (unsigned)(t + 1);
                while (__hip_atomic_load(bar, __ATOMIC_ACQUIRE,
                                         __HIP_MEMORY_SCOPE_AGENT) < target)
                    __builtin_amdgcn_s_sleep(1);
            }
            __syncthreads();
            __threadfence();          // acquire: see other blocks' h stores
        }
    }
}

// ---------------------------------------------------------------------------
extern "C" void kernel_launch(void* const* d_in, const int* in_sizes, int n_in,
                              void* d_out, int out_size, void* d_ws, size_t ws_size,
                              hipStream_t stream) {
    const float* x   = (const float*)d_in[0];   // [32][2048][1024]
    const float* Wih = (const float*)d_in[1];   // [4096][1024]
    const float* Whh = (const float*)d_in[2];   // [4096][1024]
    const float* bih = (const float*)d_in[3];   // [4096]
    const float* bhh = (const float*)d_in[4];   // [4096]
    float* out = (float*)d_out;                 // h (32*1024) then c (32*1024)

    char* ws = (char*)d_ws;
    _Float16* Wpk  = (_Float16*)(ws + OFF_W);
    float*    bias = (float*)   (ws + OFF_BIAS);
    _Float16* h0   = (_Float16*)(ws + OFF_H0);
    _Float16* h1   = (_Float16*)(ws + OFF_H1);
    unsigned* bar  = (unsigned*)(ws + OFF_BAR);

    (void)in_sizes; (void)n_in; (void)out_size; (void)ws_size;

    // Re-initialize mutable state every launch (graph-replay safe).
    hipMemsetAsync(h0, 0, (size_t)B_ * H_ * sizeof(_Float16), stream);
    hipMemsetAsync(bar, 0, 256, stream);

    pack_weights_kernel<<<2048, 256, 0, stream>>>(Wih, Whh, Wpk);
    sum_bias_kernel<<<16, 256, 0, stream>>>(bih, bhh, bias);
    lstm_persistent_kernel<<<NWG, 256, 0, stream>>>(x, Wpk, bias, h0, h1, bar, out);
}
